// EdgeModel_29137058136336
// MI455X (gfx1250) — matrix operand order repaired and checked by reading, not compiled, not run on hardware
//
#include <hip/hip_runtime.h>

typedef __attribute__((ext_vector_type(16))) _Float16 v16h;
typedef __attribute__((ext_vector_type(8)))  _Float16 v8h;
typedef __attribute__((ext_vector_type(8)))  float    v8f;
typedef __attribute__((ext_vector_type(4)))  float    v4f;

#define CIN   128
#define COUT  128
#define KONE  256
#define LEAKY 0.2f
#define LNEPS 1e-5f

// ---------------------------------------------------------------------------
// B-fragment (16-bit, 32x16 KxN) from row-major W[N][K] (f16):
// lane L holds column n = L%16; elems 0..7 -> K=c0..c0+7, elems 8..15 -> K=c0+16..c0+23
// with c0 = kb + 8*(L/16).
// ---------------------------------------------------------------------------
__device__ __forceinline__ v16h loadB(const _Float16* __restrict__ W, int K,
                                      int nb, int kb, int lane) {
  const int n  = nb + (lane & 15);
  const int k0 = kb + ((lane >> 4) << 3);
  v8h lo = *(const v8h*)(W + (size_t)n * K + k0);
  v8h hi = *(const v8h*)(W + (size_t)n * K + k0 + 16);
  v16h f;
#pragma unroll
  for (int i = 0; i < 8; ++i) { f[i] = lo[i]; f[8 + i] = hi[i]; }
  return f;
}

// ---------------------------------------------------------------------------
// A-fragment (16-bit, 16x32 MxK) from a row-major f16 row in LDS:
// lane L is row m=L%16; elems 0..7 -> K=c0..c0+7, elems 8..15 -> K=c0+16..c0+23
// with c0 = kb + 8*(L/16).
// ---------------------------------------------------------------------------
__device__ __forceinline__ v16h loadA_lds(const _Float16* hrow, int kb, int hl) {
  const int c0 = kb + hl * 8;
  v8h lo = *(const v8h*)(hrow + c0);
  v8h hi = *(const v8h*)(hrow + c0 + 16);
  v16h f;
#pragma unroll
  for (int i = 0; i < 8; ++i) { f[i] = lo[i]; f[8 + i] = hi[i]; }
  return f;
}

__global__ void convert_weights_kernel(const float* __restrict__ W1,
                                       const float* __restrict__ W2,
                                       const float* __restrict__ W3,
                                       _Float16* __restrict__ o) {
  int i = blockIdx.x * blockDim.x + threadIdx.x;          // 65536 total
  if (i < 32768)       o[i] = (_Float16)W1[i];            // 128x256
  else if (i < 49152)  o[i] = (_Float16)W2[i - 32768];    // 128x128
  else if (i < 65536)  o[i] = (_Float16)W3[i - 49152];    // 128x128
}

__global__ __launch_bounds__(256) void edge_mlp_kernel(
    const float* __restrict__ x,          // [nNodes][128]
    const int*   __restrict__ eidx,       // [2][E]  (row0=src, row1=dest)
    const _Float16* __restrict__ W1h,     // [128][256]
    const _Float16* __restrict__ W2h,     // [128][128]
    const _Float16* __restrict__ W3h,     // [128][128]
    const float* __restrict__ b1,
    const float* __restrict__ b2,
    const float* __restrict__ gamma,
    const float* __restrict__ beta,
    float* __restrict__ out,              // [E][128]
    int E, int nNodes) {
  __shared__ _Float16 hbuf[8][16][COUT];  // 32 KB, one 16x128 tile per wave

  const int tid   = threadIdx.x;
  const int wid   = tid >> 5;
  const int lane  = tid & 31;
  const int m     = lane & 15;      // row within tile (A / index role)
  const int hl    = lane >> 4;      // lane half
  const int nlo   = lane & 15;      // column within N tile (B/C role)
  const int eBase = blockIdx.x * 128 + wid * 16;

  // --- per-lane edge row indices (clamped so EXEC stays full for WMMA) ---
  int ec = eBase + m; if (ec >= E) ec = E - 1;
  int dRow = eidx[(size_t)E + ec];                  // dest -> msg[:,0:128]
  int sRow = eidx[ec];                              // src  -> msg[:,128:256]
  if ((unsigned)dRow >= (unsigned)nNodes) dRow = 0;
  if ((unsigned)sRow >= (unsigned)nNodes) sRow = 0;

  // --- layer-1 A fragments: gather x rows (f32, L2-resident) -> f16 ------
  v16h a1[8];
#pragma unroll
  for (int kt = 0; kt < 8; ++kt) {
    const int kb  = kt * 32;
    const int row = (kb < CIN) ? dRow : sRow;
    const int c0  = (kb < CIN ? kb : kb - CIN) + hl * 8;
    const float* p = x + (size_t)row * CIN + c0;    // 32B aligned
    v4f A = *(const v4f*)(p);
    v4f B = *(const v4f*)(p + 4);
    v4f C = *(const v4f*)(p + 16);
    v4f D = *(const v4f*)(p + 20);
    v16h f;
#pragma unroll
    for (int i = 0; i < 4; ++i) {
      f[i]      = (_Float16)A[i];
      f[4 + i]  = (_Float16)B[i];
      f[8 + i]  = (_Float16)C[i];
      f[12 + i] = (_Float16)D[i];
    }
    a1[kt] = f;
  }

  // --- layer 1: [16x256] @ W1^T -> [16x128], LeakyReLU, to LDS -----------
#pragma unroll
  for (int nt = 0; nt < 8; ++nt) {
    const int nb = nt * 16;
    const float bn = b1[nb + nlo];
    v8f acc = {bn, bn, bn, bn, bn, bn, bn, bn};
#pragma unroll
    for (int kt = 0; kt < 8; ++kt) {
      v16h bf = loadB(W1h, KONE, nb, kt * 32, lane);
      acc = __builtin_amdgcn_wmma_f32_16x16x32_f16(false, a1[kt], false, bf,
                                                   (short)0, acc, false, false);
    }
#pragma unroll
    for (int r = 0; r < 8; ++r) {
      float v = acc[r];
      v = (v >= 0.f) ? v : LEAKY * v;
      hbuf[wid][r + 8 * hl][nb + nlo] = (_Float16)v;  // C-layout: M=r+8*hl
    }
  }
  __syncthreads();

  // --- layer 2: [16x128] @ W2^T, LeakyReLU ------------------------------
  const _Float16* hrow = &hbuf[wid][m][0];
  v16h a2[4];
#pragma unroll
  for (int kt = 0; kt < 4; ++kt) a2[kt] = loadA_lds(hrow, kt * 32, hl);
  __syncthreads();

#pragma unroll
  for (int nt = 0; nt < 8; ++nt) {
    const int nb = nt * 16;
    const float bn = b2[nb + nlo];
    v8f acc = {bn, bn, bn, bn, bn, bn, bn, bn};
#pragma unroll
    for (int kt = 0; kt < 4; ++kt) {
      v16h bf = loadB(W2h, COUT, nb, kt * 32, lane);
      acc = __builtin_amdgcn_wmma_f32_16x16x32_f16(false, a2[kt], false, bf,
                                                   (short)0, acc, false, false);
    }
#pragma unroll
    for (int r = 0; r < 8; ++r) {
      float v = acc[r];
      v = (v >= 0.f) ? v : LEAKY * v;
      hbuf[wid][r + 8 * hl][nb + nlo] = (_Float16)v;
    }
  }
  __syncthreads();

  // --- layer 3: [16x128] @ W3^T (no bias), keep f32 tiles in registers ---
  v16h a3[4];
#pragma unroll
  for (int kt = 0; kt < 4; ++kt) a3[kt] = loadA_lds(hrow, kt * 32, hl);

  v8f res[8];
#pragma unroll
  for (int nt = 0; nt < 8; ++nt) {
    const int nb = nt * 16;
    v8f acc = {0.f, 0.f, 0.f, 0.f, 0.f, 0.f, 0.f, 0.f};
#pragma unroll
    for (int kt = 0; kt < 4; ++kt) {
      v16h bf = loadB(W3h, COUT, nb, kt * 32, lane);
      acc = __builtin_amdgcn_wmma_f32_16x16x32_f16(false, a3[kt], false, bf,
                                                   (short)0, acc, false, false);
    }
    res[nt] = acc;
  }

  // --- LayerNorm over the 128 features of each row M = r + 8*hl ----------
  // Row M is striped over the 16 lanes of one half (N = lane%16, 8 N-tiles),
  // so reduce within each 16-lane half via shfl_xor.
  float mean_[8], rs_[8];
#pragma unroll
  for (int r = 0; r < 8; ++r) {
    float s1 = 0.f, s2 = 0.f;
#pragma unroll
    for (int nt = 0; nt < 8; ++nt) {
      float v = res[nt][r];
      s1 += v;
      s2 += v * v;
    }
#pragma unroll
    for (int mk = 1; mk < 16; mk <<= 1) {
      s1 += __shfl_xor(s1, mk, 32);
      s2 += __shfl_xor(s2, mk, 32);
    }
    const float mu  = s1 * (1.0f / 128.0f);
    const float var = s2 * (1.0f / 128.0f) - mu * mu;
    mean_[r] = mu;
    rs_[r]   = rsqrtf(var + LNEPS);
  }

  // --- scale/shift + store ----------------------------------------------
#pragma unroll
  for (int nt = 0; nt < 8; ++nt) {
    const int n = nt * 16 + nlo;
    const float g  = gamma[n];
    const float bt = beta[n];
#pragma unroll
    for (int r = 0; r < 8; ++r) {
      const int eo = eBase + r + 8 * hl;
      if (eo < E) {
        float v = (res[nt][r] - mean_[r]) * rs_[r] * g + bt;
        out[(size_t)eo * COUT + n] = v;
      }
    }
  }
}

extern "C" void kernel_launch(void* const* d_in, const int* in_sizes, int n_in,
                              void* d_out, int out_size, void* d_ws, size_t ws_size,
                              hipStream_t stream) {
  (void)n_in; (void)out_size; (void)ws_size;
  const float* x     = (const float*)d_in[0];
  const int*   eidx  = (const int*)d_in[1];
  const float* W1    = (const float*)d_in[2];
  const float* b1    = (const float*)d_in[3];
  const float* W2    = (const float*)d_in[4];
  const float* b2    = (const float*)d_in[5];
  const float* W3    = (const float*)d_in[6];
  const float* gamma = (const float*)d_in[7];
  const float* beta  = (const float*)d_in[8];

  const int E      = in_sizes[1] / 2;
  const int nNodes = in_sizes[0] / CIN;

  _Float16* wh = (_Float16*)d_ws;               // 65536 f16 = 128 KB scratch
  convert_weights_kernel<<<256, 256, 0, stream>>>(W1, W2, W3, wh);

  const int blocks = (E + 127) / 128;
  edge_mlp_kernel<<<blocks, 256, 0, stream>>>(
      x, eidx, wh, wh + 32768, wh + 49152,
      b1, b2, gamma, beta, (float*)d_out, E, nNodes);
}
